// ProteinPocketEncoder_62277025792338
// MI455X (gfx1250) — matrix operand order repaired and checked by reading, not compiled, not run on hardware
//
#include <hip/hip_runtime.h>
#include <math.h>

typedef __attribute__((ext_vector_type(16))) __bf16 v16bf;
typedef __attribute__((ext_vector_type(8)))  float  v8f;
typedef __attribute__((ext_vector_type(8)))  unsigned int v8u;
typedef __attribute__((ext_vector_type(4)))  unsigned int u32x4;
typedef __attribute__((ext_vector_type(8)))  unsigned int u32x8;

#define N_  32768
#define E_  524288
#define B_  64
#define PER_ 512
#define H_  128
#define V_  64
#define NH_ 8
#define HD_ 16
#define RS_ 321           // padded radS row stride (floats)
#define WBLOB_US 61440    // per-layer swizzled weight blob, ushorts (120 tiles * 512)

// native bf16 converts (clang lowers to hardware cvt on gfx1250; RNE)
__device__ __forceinline__ unsigned short f2bf(float f) {
  __bf16 b = (__bf16)f;
  return __builtin_bit_cast(unsigned short, b);
}
__device__ __forceinline__ float bf2f(unsigned short b) {
  return (float)__builtin_bit_cast(__bf16, b);
}
// A-operand swizzle: 16x32 bf16 tile, element (m,k) -> ushort index in 512-entry tile
__device__ __forceinline__ int toffA(int m, int k) {
  int lane = m + ((((k >> 3) & 1)) << 4);
  int i = ((k >> 4) << 3) | (k & 7);
  return (lane << 4) | i;
}
// B-operand swizzle: 32x16 bf16 tile, element (k,n)
__device__ __forceinline__ int toffB(int n, int k) {
  int lane = n + ((((k >> 3) & 1)) << 4);
  int i = ((k >> 4) << 3) | (k & 7);
  return (lane << 4) | i;
}
__device__ __forceinline__ v16bf ld16(const unsigned short* p) {
  return __builtin_bit_cast(v16bf, *(const v8u*)p);
}
__device__ __forceinline__ v8f wmma_bf16(v16bf a, v16bf b, v8f c) {
  return __builtin_amdgcn_wmma_f32_16x16x32_bf16(false, a, false, b, (short)0, c, false, false);
}
__device__ __forceinline__ void atomicAddF32(float* addr, float val) {
  asm volatile("global_atomic_add_f32 %0, %1, off" :: "v"(addr), "v"(val) : "memory");
}

// ---------------- edge geometry: rhat + swizzled bf16 rbf A-tiles (K padded 16->32) -------------
__global__ __launch_bounds__(256) void geom_kernel(const float* __restrict__ pos,
    const int* __restrict__ esrc, const int* __restrict__ edst,
    float* __restrict__ rhat, unsigned short* __restrict__ rbfA) {
  int e = blockIdx.x * 256 + threadIdx.x;
  int s = esrc[e], d = edst[e];
  float dx = pos[d*3+0]-pos[s*3+0];
  float dy = pos[d*3+1]-pos[s*3+1];
  float dz = pos[d*3+2]-pos[s*3+2];
  float dist = sqrtf(dx*dx+dy*dy+dz*dz);
  float inv = 1.0f/(dist+1e-8f);
  rhat[e*3+0]=dx*inv; rhat[e*3+1]=dy*inv; rhat[e*3+2]=dz*inv;
  float fc = 0.5f*(__cosf(3.14159265358979f*fminf(dist*0.1f,1.0f))+1.0f);
  unsigned short* base = rbfA + (size_t)(e>>4)*512;
  int m = e & 15;
  for (int k=0;k<32;k++) {
    float val = 0.0f;
    if (k < 16) { float ck = 0.6666666667f*(float)k; float dd = dist-ck; val = __expf(-dd*dd*1.28f)*fc; }
    base[toffA(m,k)] = f2bf(val);
  }
}

// ---------------- one-time prep: per-layer WMMA-swizzled bf16 weight blob -----------------------
// blob layout (ushort units): [0,8*512) W1 | [8*512,88*512) W2 | [88*512,104*512) Wvs | [104*512,120*512) Wsv
__global__ __launch_bounds__(256) void prep_weights(const float* __restrict__ rad_w1,
    const float* __restrict__ rad_w2, const float* __restrict__ w_vs, const float* __restrict__ w_sv,
    unsigned short* __restrict__ wblob) {
  int l = blockIdx.y;
  unsigned short* blob = wblob + (size_t)l*WBLOB_US;
  const float* w1  = rad_w1 + (size_t)l*16*128;
  const float* w2  = rad_w2 + (size_t)l*128*320;
  const float* wvs = w_vs   + (size_t)l*64*128;
  const float* wsv = w_sv   + (size_t)l*128*64;
  int tid0 = blockIdx.x*256 + threadIdx.x;
  int stride = gridDim.x*256;
  for (int idx = tid0; idx < 32*128; idx += stride) {          // W1 (K padded 16->32)
    int k = idx >> 7, n = idx & 127;
    float val = (k < 16) ? w1[k*128 + n] : 0.0f;
    blob[(n>>4)*512 + toffB(n&15, k&31)] = f2bf(val);
  }
  for (int idx = tid0; idx < 128*320; idx += stride) {         // W2
    int k = idx / 320, n = idx % 320;
    blob[8*512 + ((k>>5)*20 + (n>>4))*512 + toffB(n&15, k&31)] = f2bf(w2[(size_t)k*320 + n]);
  }
  for (int idx = tid0; idx < 64*128; idx += stride) {          // Wvs
    int k = idx >> 7, n = idx & 127;
    blob[88*512 + ((k>>5)*8 + (n>>4))*512 + toffB(n&15, k&31)] = f2bf(wvs[k*128 + n]);
  }
  for (int idx = tid0; idx < 128*64; idx += stride) {          // Wsv
    int k = idx >> 6, n = idx & 63;
    blob[104*512 + ((k>>5)*4 + (n>>4))*512 + toffB(n&15, k&31)] = f2bf(wsv[k*64 + n]);
  }
}

// ---------------- generic WMMA GEMM: C = act(A@B + bias) [+res], 64x64 tile / 256 thr ----------
template<bool AF32, int ACT>
__global__ __launch_bounds__(256) void gemm_tile(const void* __restrict__ Aptr,
    const float* __restrict__ Bw, const float* __restrict__ bias, const float* __restrict__ Cres,
    float* __restrict__ outF, unsigned short* __restrict__ outBF,
    int M, int N, int K, int Kreal, int lda) {
  const float* Af = (const float*)Aptr;
  const unsigned short* Au = (const unsigned short*)Aptr;
  __shared__ __align__(32) unsigned short Ab[4*512];
  __shared__ __align__(32) unsigned short Bb[8*4*512];   // K up to 256
  int tid = threadIdx.x, lane = tid & 31, wv = tid >> 5;
  int row0 = blockIdx.x * 64, col0 = blockIdx.y * 64;
  for (int idx = tid; idx < K*64; idx += 256) {
    int k = idx >> 6, n = idx & 63;
    float val = (k < Kreal) ? Bw[(size_t)k*N + col0 + n] : 0.0f;
    Bb[(k>>5)*2048 + (n>>4)*512 + toffB(n&15, k&31)] = f2bf(val);
  }
  int mt0 = wv >> 2, nt = wv & 3;
  v8f acc0 = {}; v8f acc1 = {};
  int nkt = K >> 5;
  for (int kt = 0; kt < nkt; kt++) {
    __syncthreads();
    for (int idx = tid; idx < 2048; idx += 256) {
      int mm = idx >> 5, kk = idx & 31, k = kt*32 + kk;
      unsigned short bits;
      if (AF32) bits = f2bf((k < Kreal) ? Af[(size_t)(row0+mm)*lda + k] : 0.0f);
      else      bits = (k < Kreal) ? Au[(size_t)(row0+mm)*lda + k] : (unsigned short)0;
      Ab[(mm>>4)*512 + toffA(mm&15, kk)] = bits;
    }
    __syncthreads();
    v16bf b  = ld16(Bb + kt*2048 + nt*512 + lane*16);
    v16bf a0 = ld16(Ab + mt0*512 + lane*16);
    v16bf a1 = ld16(Ab + (mt0+2)*512 + lane*16);
    acc0 = wmma_bf16(a0, b, acc0);
    acc1 = wmma_bf16(a1, b, acc1);
  }
  int half = lane >> 4, n = lane & 15;
  for (int s2 = 0; s2 < 2; s2++) {
    v8f acc = s2 ? acc1 : acc0;
    int mt = s2 ? (mt0+2) : mt0;
    for (int j = 0; j < 8; j++) {
      int gr = row0 + mt*16 + j + 8*half;
      int gc = col0 + nt*16 + n;
      float val = acc[j];
      if (bias) val += bias[gc];
      if (ACT == 1) val = val > 0.0f ? val : 0.0f;
      if (Cres) val += Cres[(size_t)gr*N + gc];
      if (outF)  outF [(size_t)gr*N + gc] = val;
      if (outBF) outBF[(size_t)gr*N + gc] = f2bf(val);
    }
  }
}

// ---------------- fused per-layer edge kernel: 64 edges per workgroup ---------------------------
__global__ __launch_bounds__(256) void edge_layer(
    const int* __restrict__ esrc, const int* __restrict__ edst,
    const float* __restrict__ rhat, const unsigned short* __restrict__ rbfA,
    const unsigned short* __restrict__ s_bf, const float* __restrict__ vfeat,
    const unsigned short* __restrict__ wblobL,   // this layer's swizzled weight blob (122880 B)
    const float* __restrict__ b1, const float* __restrict__ b2,
    float* __restrict__ agg_s, float* __restrict__ agg_v) {
  extern __shared__ __align__(32) char smem[];
  unsigned short* W1b  = (unsigned short*)smem;   // 8 tiles   (TDM fills W1b..WSVb)
  unsigned short* W2b  = W1b  + 8*512;            // 80 tiles (4 kt x 20 nt)
  unsigned short* WVSb = W2b  + 80*512;           // 16 tiles (2 kt x 8 nt)
  unsigned short* WSVb = WVSb + 16*512;           // 16 tiles (4 kt x 4 nt)
  unsigned short* hidA = WSVb + 16*512;           // 16 tiles (4 rt x 4 kt)
  unsigned short* sjA  = hidA + 16*512;           // 16 tiles (4 rt x 4 kt)
  unsigned short* dvA  = sjA  + 16*512;           // 8 tiles  (4 rt x 2 kt)
  float* radS = (float*)(dvA + 8*512);            // [64][RS_]
  float* b1s  = radS + 64*RS_;                    // 128
  float* b2s  = b1s + 128;                        // 320
  float* rhS  = b2s + 320;                        // 192
  int*   dstS = (int*)(rhS + 192);                // 64
  int*   srcS = dstS + 64;                        // 64

  int tid = threadIdx.x, lane = tid & 31, wv = tid >> 5;
  int chunk = blockIdx.x, e0 = chunk * 64;
  int half = lane >> 4, nn = lane & 15;

  // ---- stage 1: TDM weight blob -> LDS[0..122880) ; biases + edge metadata by threads ----
  if (wv == 0) {
    unsigned long long ga = (unsigned long long)wblobL;
    unsigned ldsbase = __builtin_amdgcn_groupstaticsize();   // dynamic LDS starts after static (0)
    u32x4 g0;
    g0[0] = 1u;                                         // count=1, user mode
    g0[1] = ldsbase;                                    // lds_addr
    g0[2] = (unsigned)(ga & 0xffffffffu);               // global_addr[31:0]
    g0[3] = (unsigned)((ga >> 32) & 0x1ffffffu) | 0x80000000u;  // addr[56:32] | type=2
    u32x8 g1;
    g1[0] = 3u << 16;                                   // data_size = 8 bytes
    g1[1] = (15360u & 0xffffu) << 16;                   // tensor_dim0 lo16  (122880/8)
    g1[2] = 0u;                                         // tensor_dim0 hi16, tensor_dim1 lo16
    g1[3] = (15360u & 0xffffu) << 16;                   // tile_dim0
    g1[4] = 0u;                                         // tile_dim1=0 (unused), tile_dim2=0
    g1[5] = 15360u;                                     // tensor_dim0_stride lo32
    g1[6] = 0u;
    g1[7] = 0u;
    asm volatile("tensor_load_to_lds %0, %1" :: "s"(g0), "s"(g1) : "memory");
  }
  if (tid < 128) b1s[tid] = b1[tid];
  for (int idx = tid; idx < 320; idx += 256) b2s[idx] = b2[idx];
  for (int idx = tid; idx < 192; idx += 256) rhS[idx] = rhat[(size_t)e0*3 + idx];
  if (tid < 64) {
    int sv = esrc[e0 + tid], dvn = edst[e0 + tid];
    srcS[tid] = sv; dstS[tid] = dvn;
    __builtin_prefetch(s_bf + (size_t)sv*128, 0, 0);     // warm gather lines
    __builtin_prefetch(vfeat + (size_t)sv*192, 0, 0);
    __builtin_prefetch(agg_s + (size_t)dvn*128, 1, 0);
  }
  if (wv == 0) __builtin_amdgcn_s_wait_tensorcnt(0);
  __syncthreads();

  // ---- GEMM1: hidden = silu(rbf @ w1 + b1)  [64 x 128] ----
  for (int t = wv; t < 32; t += 8) {
    int rt = t >> 3, ct = t & 7;
    v16bf a = __builtin_bit_cast(v16bf, *(const v8u*)(rbfA + ((size_t)(chunk*4 + rt))*512 + lane*16));
    v16bf b = ld16(W1b + ct*512 + lane*16);
    v8f acc = {};
    acc = wmma_bf16(a, b, acc);
    for (int j = 0; j < 8; j++) {
      int mloc = j + 8*half;
      int c = ct*16 + nn;
      float h = acc[j] + b1s[c];
      h = h * __builtin_amdgcn_rcpf(1.0f + __expf(-h));   // silu, fast rcp
      hidA[(rt*4 + (c>>5))*512 + toffA(mloc, c&31)] = f2bf(h);
    }
  }
  __syncthreads();

  // ---- GEMM2: rad = hidden @ w2 + b2  [64 x 320] -> radS f32 ----
  for (int t = wv; t < 80; t += 8) {
    int rt = t / 20, ct = t % 20;
    v8f acc = {};
    for (int kt = 0; kt < 4; kt++) {
      v16bf a = ld16(hidA + (rt*4 + kt)*512 + lane*16);
      v16bf b = ld16(W2b + (kt*20 + ct)*512 + lane*16);
      acc = wmma_bf16(a, b, acc);
    }
    for (int j = 0; j < 8; j++) {
      int rl = rt*16 + j + 8*half;
      int c = ct*16 + nn;
      radS[rl*RS_ + c] = acc[j] + b2s[c];
    }
  }
  __syncthreads();

  // ---- gather stage: s_j -> sjA, (g_vs*dotv) -> dvA (4 threads per edge) ----
  {
    int el = tid >> 2, q = tid & 3;
    int sn = srcS[el];
    int rt_e = el >> 4, m_e = el & 15;
    for (int i = 0; i < 32; i++) {
      int c = q*32 + i;
      sjA[(rt_e*4 + (c>>5))*512 + toffA(m_e, c&31)] = s_bf[(size_t)sn*128 + c];
    }
    float r0 = rhS[el*3+0], r1 = rhS[el*3+1], r2 = rhS[el*3+2];
    for (int i = 0; i < 16; i++) {
      int ch = q*16 + i;
      float dv = vfeat[((size_t)sn*3+0)*64 + ch]*r0
               + vfeat[((size_t)sn*3+1)*64 + ch]*r1
               + vfeat[((size_t)sn*3+2)*64 + ch]*r2;
      float g = radS[el*RS_ + 128 + ch];        // g_vs
      dvA[(rt_e*2 + (ch>>5))*512 + toffA(m_e, ch&31)] = f2bf(g*dv);
    }
  }
  __syncthreads();

  // ---- GEMM3: msg_s = g_ss*s_j + (g_vs*dotv)@w_vs ; atomic scatter ----
  for (int t = wv; t < 32; t += 8) {
    int rt = t >> 3, ct = t & 7;
    v8f acc = {};
    for (int kt = 0; kt < 2; kt++) {
      v16bf a = ld16(dvA + (rt*2 + kt)*512 + lane*16);
      v16bf b = ld16(WVSb + (kt*8 + ct)*512 + lane*16);
      acc = wmma_bf16(a, b, acc);
    }
    for (int j = 0; j < 8; j++) {
      int rl = rt*16 + j + 8*half;
      int c = ct*16 + nn;
      float sj = bf2f(sjA[(rt*4 + (c>>5))*512 + toffA(rl&15, c&31)]);
      float msg = radS[rl*RS_ + c]*sj + acc[j];
      atomicAddF32(agg_s + (size_t)dstS[rl]*128 + c, msg);
    }
  }

  // ---- GEMM4: sv = (s_j@w_sv)*g_sv ; msg_v = g_vv*v_j + sv*rhat ; atomic scatter ----
  for (int t = wv; t < 16; t += 8) {
    int rt = t >> 2, ct = t & 3;
    v8f acc = {};
    for (int kt = 0; kt < 4; kt++) {
      v16bf a = ld16(sjA + (rt*4 + kt)*512 + lane*16);
      v16bf b = ld16(WSVb + (kt*4 + ct)*512 + lane*16);
      acc = wmma_bf16(a, b, acc);
    }
    for (int j = 0; j < 8; j++) {
      int rl = rt*16 + j + 8*half;
      int ch = ct*16 + nn;
      float svv = acc[j] * radS[rl*RS_ + 192 + ch];   // g_sv
      float gvv = radS[rl*RS_ + 256 + ch];            // g_vv
      int sn = srcS[rl], dn = dstS[rl];
      for (int k = 0; k < 3; k++) {
        float vj = vfeat[((size_t)sn*3 + k)*64 + ch];
        float m = gvv*vj + svv*rhS[rl*3 + k];
        atomicAddF32(agg_v + ((size_t)dn*3 + k)*64 + ch, m);
      }
    }
  }
}

// ---------------- attention: one (qtile, head, pocket) per workgroup ---------------------------
__global__ __launch_bounds__(256) void attn_kernel(const unsigned short* __restrict__ qkv,
                                                   unsigned short* __restrict__ ao) {
  extern __shared__ __align__(32) char smem[];
  unsigned short* Qa = (unsigned short*)smem;   // 4 tiles
  unsigned short* Kb = Qa + 4*512;              // 32 tiles
  unsigned short* Vb = Kb + 32*512;             // 16 tiles
  unsigned short* Pa = Vb + 16*512;             // 64 tiles (4 rt x 16 kt)
  float* S    = (float*)(Pa + 64*512);          // [64][516]
  float* pm   = S + 64*516;                     // [64][4]
  float* rsum = pm + 256;                       // [64]

  int tid = threadIdx.x, lane = tid & 31, wv = tid >> 5;
  int qt = blockIdx.x, head = blockIdx.y, pocket = blockIdx.z;
  int nodeQ0 = pocket*PER_ + qt*64;
  int nodeK0 = pocket*PER_;
  int half = lane >> 4, nnn = lane & 15;

  for (int idx = tid; idx < 64*32; idx += 256) {       // Q (pre-scaled by 1/sqrt(HD)=0.25)
    int m = idx >> 5, d = idx & 31;
    unsigned short bits = 0;
    if (d < 16) bits = f2bf(bf2f(qkv[(size_t)(nodeQ0+m)*384 + head*16 + d]) * 0.25f);
    Qa[(m>>4)*512 + toffA(m&15, d)] = bits;
  }
  for (int idx = tid; idx < 512*32; idx += 256) {      // K^T (B operand, contraction=d)
    int key = idx >> 5, d = idx & 31;
    unsigned short bits = (d < 16) ? qkv[(size_t)(nodeK0+key)*384 + 128 + head*16 + d] : (unsigned short)0;
    Kb[(key>>4)*512 + toffB(key&15, d)] = bits;
  }
  for (int idx = tid; idx < 512*16; idx += 256) {      // V (B operand, contraction=key)
    int key = idx >> 4, d = idx & 15;
    Vb[(key>>5)*512 + toffB(d, key&31)] = qkv[(size_t)(nodeK0+key)*384 + 256 + head*16 + d];
  }
  __syncthreads();

  // S = Q @ K^T  (128 tiles)
  for (int t = wv; t < 128; t += 8) {
    int rt = t >> 5, ktile = t & 31;
    v16bf a = ld16(Qa + rt*512 + lane*16);
    v16bf b = ld16(Kb + ktile*512 + lane*16);
    v8f acc = {};
    acc = wmma_bf16(a, b, acc);
    for (int j = 0; j < 8; j++) {
      int r = rt*16 + j + 8*half;
      int c = ktile*16 + nnn;
      S[r*516 + c] = acc[j];
    }
  }
  __syncthreads();

  // softmax over 512 keys (4 threads per row)
  {
    int r = tid >> 2, p = tid & 3;
    float mx = -3.4e38f;
    for (int c = p*128; c < p*128 + 128; c++) mx = fmaxf(mx, S[r*516 + c]);
    pm[r*4 + p] = mx;
    __syncthreads();
    float mxr = fmaxf(fmaxf(pm[r*4], pm[r*4+1]), fmaxf(pm[r*4+2], pm[r*4+3]));
    float sum = 0.0f;
    for (int c = p*128; c < p*128 + 128; c++) {
      float pv = __expf(S[r*516 + c] - mxr);
      sum += pv;
      Pa[((r>>4)*16 + (c>>5))*512 + toffA(r&15, c&31)] = f2bf(pv);
    }
    __syncthreads();
    pm[r*4 + p] = sum;
    __syncthreads();
    if (p == 0) rsum[r] = pm[r*4] + pm[r*4+1] + pm[r*4+2] + pm[r*4+3];
  }
  __syncthreads();

  // O = (P @ V) * (1/rowsum)  (4 tiles, waves 0..3)
  if (wv < 4) {
    int rt = wv;
    v8f acc = {};
    for (int kt = 0; kt < 16; kt++) {
      v16bf a = ld16(Pa + (rt*16 + kt)*512 + lane*16);
      v16bf b = ld16(Vb + kt*512 + lane*16);
      acc = wmma_bf16(a, b, acc);
    }
    for (int j = 0; j < 8; j++) {
      int r = rt*16 + j + 8*half;
      float o = acc[j] * __builtin_amdgcn_rcpf(rsum[r]);
      ao[(size_t)(nodeQ0 + r)*128 + head*16 + nnn] = f2bf(o);
    }
  }
}

// ---------------- mean pool per pocket ----------------------------------------------------------
__global__ __launch_bounds__(128) void pool_kernel(const float* __restrict__ ao2,
                                                   float* __restrict__ pooled) {
  int b = blockIdx.x, c = threadIdx.x;
  float s = 0.0f;
  for (int i = 0; i < PER_; i++) s += ao2[(size_t)(b*PER_ + i)*128 + c];
  pooled[b*128 + c] = s * (1.0f/512.0f);
}

// ================================================================================================
extern "C" void kernel_launch(void* const* d_in, const int* in_sizes, int n_in,
                              void* d_out, int out_size, void* d_ws, size_t ws_size,
                              hipStream_t stream) {
  (void)in_sizes; (void)n_in; (void)out_size; (void)ws_size;
  const float* x      = (const float*)d_in[0];
  const float* pos    = (const float*)d_in[1];
  const int*   eidx   = (const int*)d_in[2];
  const float* node_w = (const float*)d_in[4];
  const float* node_b = (const float*)d_in[5];
  const float* rad_w1 = (const float*)d_in[6];
  const float* rad_b1 = (const float*)d_in[7];
  const float* rad_w2 = (const float*)d_in[8];
  const float* rad_b2 = (const float*)d_in[9];
  const float* w_vs   = (const float*)d_in[10];
  const float* w_sv   = (const float*)d_in[11];
  const float* w_upd  = (const float*)d_in[12];
  const float* b_upd  = (const float*)d_in[13];
  const float* w_vu   = (const float*)d_in[14];
  const float* e3_w   = (const float*)d_in[15];
  const float* e3_b   = (const float*)d_in[16];
  const float* ain_w  = (const float*)d_in[17];
  const float* ain_b  = (const float*)d_in[18];
  const float* aout_w = (const float*)d_in[19];
  const float* aout_b = (const float*)d_in[20];
  const float* pw1    = (const float*)d_in[21];
  const float* pb1    = (const float*)d_in[22];
  const float* pw2    = (const float*)d_in[23];
  const float* pb2    = (const float*)d_in[24];

  char* base = (char*)d_ws; size_t off = 0;
  auto alloc = [&](size_t bytes) -> char* {
    off = (off + 255) & ~(size_t)255;
    char* p = base + off; off += bytes; return p;
  };
  float* sF            = (float*)alloc((size_t)N_*H_*4);
  unsigned short* sB   = (unsigned short*)alloc((size_t)N_*H_*2);
  float* vF            = (float*)alloc((size_t)N_*3*V_*4);
  float* aggS          = (float*)alloc((size_t)N_*H_*4);
  float* aggV          = (float*)alloc((size_t)N_*3*V_*4);
  float* rh            = (float*)alloc((size_t)E_*3*4);
  unsigned short* rbfA = (unsigned short*)alloc((size_t)E_*32*2);
  unsigned short* hB   = (unsigned short*)alloc((size_t)N_*H_*2);
  unsigned short* qkvB = (unsigned short*)alloc((size_t)N_*384*2);
  unsigned short* aoB  = (unsigned short*)alloc((size_t)N_*H_*2);
  float* ao2           = (float*)alloc((size_t)N_*H_*4);
  float* pooled        = (float*)alloc(64*128*4);
  unsigned short* t2B  = (unsigned short*)alloc(64*256*2);
  unsigned short* wblob= (unsigned short*)alloc((size_t)4*WBLOB_US*2);

  const int* esrc = eidx;
  const int* edst = eidx + E_;

  constexpr unsigned EDGE_LDS = 160*1024 + (64*RS_ + 128 + 320 + 192)*4 + 64*4*2; // 249088
  constexpr unsigned ATTN_LDS = 116*1024 + (64*516 + 256 + 64)*4;                 // 252160

  hipMemsetAsync(vF, 0, (size_t)N_*3*V_*4, stream);
  prep_weights<<<dim3(16, 4), 256, 0, stream>>>(rad_w1, rad_w2, w_vs, w_sv, wblob);
  geom_kernel<<<dim3(E_/256), 256, 0, stream>>>(pos, esrc, edst, rh, rbfA);
  // embed: s = x @ node_w + node_b  (K padded 20->32)
  gemm_tile<true,0><<<dim3(N_/64, H_/64), 256, 0, stream>>>(
      x, node_w, node_b, nullptr, sF, sB, N_, H_, 32, 20, 20);

  for (int l = 0; l < 4; l++) {
    hipMemsetAsync(aggS, 0, (size_t)N_*H_*4, stream);
    hipMemsetAsync(aggV, 0, (size_t)N_*3*V_*4, stream);
    edge_layer<<<dim3(E_/64), 256, EDGE_LDS, stream>>>(
        esrc, edst, rh, rbfA, sB, vF,
        wblob + (size_t)l*WBLOB_US,
        rad_b1 + l*128, rad_b2 + l*320,
        aggS, aggV);
    // s = s + agg_s @ w_upd + b_upd
    gemm_tile<true,0><<<dim3(N_/64, H_/64), 256, 0, stream>>>(
        aggS, w_upd + (size_t)l*128*128, b_upd + l*128, sF, sF, sB, N_, H_, 128, 128, 128);
    // v = v + agg_v @ w_vu  (as [3N,64]@[64,64])
    gemm_tile<true,0><<<dim3(3*N_/64, 1), 256, 0, stream>>>(
        aggV, w_vu + (size_t)l*64*64, nullptr, vF, vF, nullptr, 3*N_, 64, 64, 64, 64);
  }

  // h = s @ e3_out_w + e3_out_b
  gemm_tile<false,0><<<dim3(N_/64, 2), 256, 0, stream>>>(
      sB, e3_w, e3_b, nullptr, nullptr, hB, N_, 128, 128, 128, 128);
  // qkv = h @ attn_in_w + attn_in_b
  gemm_tile<false,0><<<dim3(N_/64, 6), 256, 0, stream>>>(
      hB, ain_w, ain_b, nullptr, nullptr, qkvB, N_, 384, 128, 128, 128);
  attn_kernel<<<dim3(PER_/64, NH_, B_), 256, ATTN_LDS, stream>>>(qkvB, aoB);
  // o = o @ attn_out_w + attn_out_b
  gemm_tile<false,0><<<dim3(N_/64, 2), 256, 0, stream>>>(
      aoB, aout_w, aout_b, nullptr, ao2, nullptr, N_, 128, 128, 128, 128);
  pool_kernel<<<dim3(B_), 128, 0, stream>>>(ao2, pooled);
  // relu(pooled@proj_w1+b1) @ proj_w2 + b2
  gemm_tile<true,1><<<dim3(1, 4), 256, 0, stream>>>(
      pooled, pw1, pb1, nullptr, nullptr, t2B, 64, 256, 128, 128, 128);
  gemm_tile<false,0><<<dim3(1, 4), 256, 0, stream>>>(
      t2B, pw2, pb2, nullptr, (float*)d_out, nullptr, 64, 256, 256, 256, 256);
}
